// CausalMultiHeadAttention_62251255988856
// MI455X (gfx1250) — matrix operand order repaired and checked by reading, not compiled, not run on hardware
//
#include <hip/hip_runtime.h>

typedef __bf16 bf16_t;
typedef __bf16 v8bf  __attribute__((ext_vector_type(8)));
typedef __bf16 v16bf __attribute__((ext_vector_type(16)));
typedef float  v8f   __attribute__((ext_vector_type(8)));

#define S_LEN   4096
#define D_MODEL 1024
#define NHEAD   16
#define HD      64

// ---------------------------------------------------------------------------
// Fragment loaders for v_wmma_f32_16x16x32_bf16 (wave32).
// A (16x32, MxK): lane<16 -> row=lane, K 0..7 (v0-3) & 16..23 (v4-7);
//                 lane>=16 -> same row, K 8..15 & 24..31.
// B (32x16, KxN): lane&15 -> column, lane<16 holds K 0..7 & 16..23, lane>=16 K 8..15 & 24..31.
// C/D (16x16 f32): VGPR r, lane l: row = r + 8*(l>=16), col = l&15.
// ---------------------------------------------------------------------------
__device__ __forceinline__ v16bf load_a_frag(const bf16_t* A, int ld,
                                             int m0, int k0, int lane) {
  const int r = lane & 15, half = lane >> 4;
  const bf16_t* p = A + (size_t)(m0 + r) * ld + k0 + half * 8;
  v8bf lo = *(const v8bf*)p;          // K = k0 + 8*half + [0..7]
  v8bf hi = *(const v8bf*)(p + 16);   // K = k0 + 16 + 8*half + [0..7]
  v16bf out;
#pragma unroll
  for (int i = 0; i < 8; ++i) { out[i] = lo[i]; out[i + 8] = hi[i]; }
  return out;
}

__device__ __forceinline__ v16bf load_b_frag(const bf16_t* B, int ld,
                                             int n0, int k0, int lane) {
  const int c = lane & 15, half = lane >> 4;
  const bf16_t* p = B + (size_t)(n0 + c) * ld + k0 + half * 8;
  v8bf lo = *(const v8bf*)p;
  v8bf hi = *(const v8bf*)(p + 16);
  v16bf out;
#pragma unroll
  for (int i = 0; i < 8; ++i) { out[i] = lo[i]; out[i + 8] = hi[i]; }
  return out;
}

__device__ __forceinline__ v8f wmma_bf16(v16bf a, v16bf b, v8f c) {
  return __builtin_amdgcn_wmma_f32_16x16x32_bf16(false, a, false, b,
                                                 (short)0, c, false, false);
}

__device__ __forceinline__ v8f v8f_zero() {
  v8f z = {0.f, 0.f, 0.f, 0.f, 0.f, 0.f, 0.f, 0.f};
  return z;
}

// C[32x64] tile of A[MxK] * W[NxK]^T (both bf16 row-major), fp32 accumulate.
// Each B fragment is reused by two A fragments (rows m0..m0+15, m0+16..m0+31).
__device__ __forceinline__ void gemm_tile_32x64(const bf16_t* __restrict__ A,
                                                const bf16_t* __restrict__ W,
                                                int K, int m0, int n0, int lane,
                                                v8f acc[2][4]) {
#pragma unroll
  for (int qt = 0; qt < 2; ++qt)
#pragma unroll
    for (int t = 0; t < 4; ++t) acc[qt][t] = v8f_zero();
  for (int k0 = 0; k0 < K; k0 += 32) {
    v16bf a0 = load_a_frag(A, K, m0, k0, lane);
    v16bf a1 = load_a_frag(A, K, m0 + 16, k0, lane);
#pragma unroll
    for (int t = 0; t < 4; ++t) {
      v16bf b = load_b_frag(W, K, n0 + 16 * t, k0, lane);
      acc[0][t] = wmma_bf16(a0, b, acc[0][t]);
      acc[1][t] = wmma_bf16(a1, b, acc[1][t]);
    }
  }
}

// ---------------------------------------------------------------------------
// Kernel 1: fp32 -> bf16 conversion
// ---------------------------------------------------------------------------
__global__ void cvt_f32_bf16_kernel(const float* __restrict__ in,
                                    bf16_t* __restrict__ out, int n) {
  int i = blockIdx.x * blockDim.x + threadIdx.x;
  if (i < n) out[i] = (bf16_t)in[i];
}

// ---------------------------------------------------------------------------
// Kernel 2: fused QKV projection. z=0 -> Q (row major), z=1 -> K (row major),
// z=2 -> V stored transposed as Vt[D][S].
// ---------------------------------------------------------------------------
__global__ __launch_bounds__(32)
void qkv_proj_kernel(const bf16_t* __restrict__ xb,
                     const bf16_t* __restrict__ Wqb,
                     const bf16_t* __restrict__ Wkb,
                     const bf16_t* __restrict__ Wvb,
                     bf16_t* __restrict__ Qb, bf16_t* __restrict__ Kb,
                     bf16_t* __restrict__ Vt) {
  const int lane = threadIdx.x;
  const int m0 = blockIdx.x * 32;
  const int n0 = blockIdx.y * 64;
  const int z  = blockIdx.z;
  const bf16_t* W = (z == 0) ? Wqb : (z == 1) ? Wkb : Wvb;

  v8f acc[2][4];
  gemm_tile_32x64(xb, W, D_MODEL, m0, n0, lane, acc);

  const int r_off = (lane >> 4) * 8;
  const int c = lane & 15;
  if (z < 2) {
    bf16_t* O = (z == 0) ? Qb : Kb;
#pragma unroll
    for (int qt = 0; qt < 2; ++qt)
#pragma unroll
      for (int t = 0; t < 4; ++t)
#pragma unroll
        for (int r = 0; r < 8; ++r)
          O[(size_t)(m0 + qt * 16 + r + r_off) * D_MODEL + n0 + 16 * t + c] =
              (bf16_t)acc[qt][t][r];
  } else {
#pragma unroll
    for (int qt = 0; qt < 2; ++qt)
#pragma unroll
      for (int t = 0; t < 4; ++t)
#pragma unroll
        for (int r = 0; r < 8; ++r)
          Vt[(size_t)(n0 + 16 * t + c) * S_LEN + m0 + qt * 16 + r + r_off] =
              (bf16_t)acc[qt][t][r];
  }
}

// ---------------------------------------------------------------------------
// Kernel 3: RoPE applied in place on bf16 Q and K (fp32 math).
// ---------------------------------------------------------------------------
__global__ void rope_kernel(bf16_t* __restrict__ Qb, bf16_t* __restrict__ Kb,
                            const int* __restrict__ pos) {
  int idx = blockIdx.x * blockDim.x + threadIdx.x;   // over S * D/2 pairs
  if (idx >= S_LEN * (D_MODEL / 2)) return;
  int s = idx >> 9;          // / 512
  int p = idx & 511;
  int i = p & 31;            // pair index within head (0..31)
  int h = p >> 5;            // head
  int col = h * HD + 2 * i;
  // inv_freq = 10000^(-(2i)/64) = exp(-(i/32) * ln(10000))
  float inv = __expf(-(float)i * 0.03125f * 9.210340372f);
  float ang = (float)pos[s] * inv;
  float cs = __cosf(ang), sn = __sinf(ang);
  size_t base = (size_t)s * D_MODEL + col;
  float qe = (float)Qb[base], qo = (float)Qb[base + 1];
  Qb[base]     = (bf16_t)(qe * cs - qo * sn);
  Qb[base + 1] = (bf16_t)(qe * sn + qo * cs);
  float ke = (float)Kb[base], ko = (float)Kb[base + 1];
  Kb[base]     = (bf16_t)(ke * cs - ko * sn);
  Kb[base + 1] = (bf16_t)(ke * sn + ko * cs);
}

// ---------------------------------------------------------------------------
// Kernel 4: causal flash attention. One wave handles a 32-row query tile for
// one head; loops over 32-key kv tiles. K/V fragment loads and the LDS P
// bounce are shared across the two 16-row halves: 16 WMMAs per kv tile.
// ---------------------------------------------------------------------------
__global__ __launch_bounds__(32)
void flash_attn_kernel(const bf16_t* __restrict__ Qb,
                       const bf16_t* __restrict__ Kb,
                       const bf16_t* __restrict__ Vt,
                       bf16_t* __restrict__ Ob) {
  __shared__ __align__(32) bf16_t pbuf[32 * 32];  // P tile, 2 KB
  const int lane = threadIdx.x;
  const int q0 = blockIdx.x * 32;
  const int h  = blockIdx.y;
  const int c = lane & 15;
  const int r_off = (lane >> 4) * 8;

  // Q fragments: 32 rows x 64 head-dim -> 2 row-halves x 2 K=32 chunks.
  v16bf qa[2][2];
#pragma unroll
  for (int qt = 0; qt < 2; ++qt) {
    qa[qt][0] = load_a_frag(Qb + h * HD, D_MODEL, q0 + qt * 16, 0, lane);
    qa[qt][1] = load_a_frag(Qb + h * HD, D_MODEL, q0 + qt * 16, 32, lane);
  }

  float mrow[2][8], lrow[2][8];
  v8f acc[2][4];
#pragma unroll
  for (int qt = 0; qt < 2; ++qt) {
#pragma unroll
    for (int r = 0; r < 8; ++r) { mrow[qt][r] = -1e30f; lrow[qt][r] = 0.0f; }
#pragma unroll
    for (int t = 0; t < 4; ++t) acc[qt][t] = v8f_zero();
  }

  const int kv_end = q0 + 32;   // exclusive: need keys <= q0+31
  for (int kv = 0; kv < kv_end; kv += 32) {
    // ---- K fragments, shared by both q halves ----
    v16bf bk[2][2];
#pragma unroll
    for (int t = 0; t < 2; ++t) {
      bk[t][0] = load_b_frag(Kb + h * HD, D_MODEL, kv + 16 * t, 0, lane);
      bk[t][1] = load_b_frag(Kb + h * HD, D_MODEL, kv + 16 * t, 32, lane);
    }
    // ---- scores S = (Q K^T) * scale, 32x32 as 2x2 16x16 fragments ----
    v8f sfr[2][2];
#pragma unroll
    for (int qt = 0; qt < 2; ++qt)
#pragma unroll
      for (int t = 0; t < 2; ++t) {
        v8f s = v8f_zero();
        s = wmma_bf16(qa[qt][0], bk[t][0], s);
        s = wmma_bf16(qa[qt][1], bk[t][1], s);
        sfr[qt][t] = s;
      }
    // ---- scale + causal mask ----
#pragma unroll
    for (int qt = 0; qt < 2; ++qt)
#pragma unroll
      for (int t = 0; t < 2; ++t) {
        const int kg = kv + 16 * t + c;
#pragma unroll
        for (int r = 0; r < 8; ++r) {
          float v = sfr[qt][t][r] * 0.125f;     // 1/sqrt(64)
          if (kg > q0 + qt * 16 + r + r_off) v = -1e30f;
          sfr[qt][t][r] = v;
        }
      }
    __syncthreads();   // pbuf free (previous iteration's pa consumed)
    // ---- online softmax (rows live in 16-lane halves) ----
#pragma unroll
    for (int qt = 0; qt < 2; ++qt) {
      float mnew[8], alpha[8], rsum[8];
#pragma unroll
      for (int r = 0; r < 8; ++r) {
        float mx = fmaxf(sfr[qt][0][r], sfr[qt][1][r]);
#pragma unroll
        for (int o = 1; o < 16; o <<= 1) mx = fmaxf(mx, __shfl_xor(mx, o, 32));
        mnew[r] = fmaxf(mrow[qt][r], mx);
        alpha[r] = __expf(mrow[qt][r] - mnew[r]);
        rsum[r] = 0.0f;
      }
#pragma unroll
      for (int t = 0; t < 2; ++t)
#pragma unroll
        for (int r = 0; r < 8; ++r) {
          float p = __expf(sfr[qt][t][r] - mnew[r]);
          rsum[r] += p;
          pbuf[(qt * 16 + r + r_off) * 32 + 16 * t + c] = (bf16_t)p;
        }
#pragma unroll
      for (int r = 0; r < 8; ++r) {
#pragma unroll
        for (int o = 1; o < 16; o <<= 1) rsum[r] += __shfl_xor(rsum[r], o, 32);
        lrow[qt][r] = lrow[qt][r] * alpha[r] + rsum[r];
        mrow[qt][r] = mnew[r];
#pragma unroll
        for (int t = 0; t < 4; ++t) acc[qt][t][r] *= alpha[r];
      }
    }
    __syncthreads();
    // ---- re-layout P (32x32) as A fragments from LDS, then P.V ----
    v16bf pa0 = load_a_frag(pbuf, 32, 0, 0, lane);
    v16bf pa1 = load_a_frag(pbuf, 32, 16, 0, lane);
#pragma unroll
    for (int t = 0; t < 4; ++t) {
      v16bf bv = load_b_frag(Vt, S_LEN, h * HD + 16 * t, kv, lane);
      acc[0][t] = wmma_bf16(pa0, bv, acc[0][t]);
      acc[1][t] = wmma_bf16(pa1, bv, acc[1][t]);
    }
  }

  // ---- epilogue: normalize and store bf16 O ----
#pragma unroll
  for (int qt = 0; qt < 2; ++qt)
#pragma unroll
    for (int r = 0; r < 8; ++r) {
      float inv = 1.0f / lrow[qt][r];
#pragma unroll
      for (int t = 0; t < 4; ++t)
        Ob[(size_t)(q0 + qt * 16 + r + r_off) * D_MODEL + h * HD + 16 * t + c] =
            (bf16_t)(acc[qt][t][r] * inv);
    }
}

// ---------------------------------------------------------------------------
// Kernel 5: output projection, fp32 result into d_out.
// ---------------------------------------------------------------------------
__global__ __launch_bounds__(32)
void out_proj_kernel(const bf16_t* __restrict__ Ob,
                     const bf16_t* __restrict__ Wob,
                     float* __restrict__ out) {
  const int lane = threadIdx.x;
  const int m0 = blockIdx.x * 32;
  const int n0 = blockIdx.y * 64;
  v8f acc[2][4];
  gemm_tile_32x64(Ob, Wob, D_MODEL, m0, n0, lane, acc);
  const int r_off = (lane >> 4) * 8;
  const int c = lane & 15;
#pragma unroll
  for (int qt = 0; qt < 2; ++qt)
#pragma unroll
    for (int t = 0; t < 4; ++t)
#pragma unroll
      for (int r = 0; r < 8; ++r)
        out[(size_t)(m0 + qt * 16 + r + r_off) * D_MODEL + n0 + 16 * t + c] =
            acc[qt][t][r];
}

// ---------------------------------------------------------------------------
extern "C" void kernel_launch(void* const* d_in, const int* in_sizes, int n_in,
                              void* d_out, int out_size, void* d_ws,
                              size_t ws_size, hipStream_t stream) {
  (void)in_sizes; (void)n_in; (void)out_size; (void)ws_size;
  const float* x  = (const float*)d_in[0];
  const int*   tp = (const int*)d_in[1];
  const float* Wq = (const float*)d_in[2];
  const float* Wk = (const float*)d_in[3];
  const float* Wv = (const float*)d_in[4];
  const float* Wo = (const float*)d_in[5];
  float* out = (float*)d_out;

  char* ws = (char*)d_ws;
  const size_t MB = 1024 * 1024;
  bf16_t* xb  = (bf16_t*)(ws + 0 * MB);     // 8 MB
  bf16_t* Wqb = (bf16_t*)(ws + 8 * MB);     // 2 MB
  bf16_t* Wkb = (bf16_t*)(ws + 10 * MB);    // 2 MB
  bf16_t* Wvb = (bf16_t*)(ws + 12 * MB);    // 2 MB
  bf16_t* Wob = (bf16_t*)(ws + 14 * MB);    // 2 MB
  bf16_t* Qb  = (bf16_t*)(ws + 16 * MB);    // 8 MB
  bf16_t* Kb  = (bf16_t*)(ws + 24 * MB);    // 8 MB
  bf16_t* Vt  = (bf16_t*)(ws + 32 * MB);    // 8 MB (transposed D x S)
  bf16_t* Ob  = (bf16_t*)(ws + 40 * MB);    // 8 MB

  const int NX = S_LEN * D_MODEL;
  const int NW = D_MODEL * D_MODEL;
  cvt_f32_bf16_kernel<<<(NX + 255) / 256, 256, 0, stream>>>(x, xb, NX);
  cvt_f32_bf16_kernel<<<(NW + 255) / 256, 256, 0, stream>>>(Wq, Wqb, NW);
  cvt_f32_bf16_kernel<<<(NW + 255) / 256, 256, 0, stream>>>(Wk, Wkb, NW);
  cvt_f32_bf16_kernel<<<(NW + 255) / 256, 256, 0, stream>>>(Wv, Wvb, NW);
  cvt_f32_bf16_kernel<<<(NW + 255) / 256, 256, 0, stream>>>(Wo, Wob, NW);

  dim3 gqkv(S_LEN / 32, D_MODEL / 64, 3);
  qkv_proj_kernel<<<gqkv, 32, 0, stream>>>(xb, Wqb, Wkb, Wvb, Qb, Kb, Vt);

  const int NR = S_LEN * (D_MODEL / 2);
  rope_kernel<<<(NR + 255) / 256, 256, 0, stream>>>(Qb, Kb, tp);

  dim3 gfa(S_LEN / 32, NHEAD);
  flash_attn_kernel<<<gfa, 32, 0, stream>>>(Qb, Kb, Vt, Ob);

  dim3 gout(S_LEN / 32, D_MODEL / 64);
  out_proj_kernel<<<gout, 32, 0, stream>>>(Ob, Wob, out);
}
